// VectorQuantizer_15126874816810
// MI455X (gfx1250) — matrix operand order repaired and checked by reading, not compiled, not run on hardware
//
#include <hip/hip_runtime.h>
#include <hip/hip_bf16.h>
#include <float.h>

typedef __bf16 bf16_t;
typedef bf16_t v16bf __attribute__((ext_vector_type(16)));
typedef bf16_t v8bf  __attribute__((ext_vector_type(8)));
typedef float  v8f   __attribute__((ext_vector_type(8)));
typedef float  v4f   __attribute__((ext_vector_type(4)));
typedef int    v4i   __attribute__((ext_vector_type(4)));

#define DIMD 256
#define DIMK 1024
#define NROWS 65536
#define WAVES_PER_WG 8
#define ROWS_PER_WG (WAVES_PER_WG * 16)      // 128
#define NUM_WG (NROWS / ROWS_PER_WG)         // 512
#define NUM_WAVES (NUM_WG * WAVES_PER_WG)    // 4096

// LDS staging of B k-tiles: 16 rows x 512B, padded to 528B/row so the 16 lanes
// of a half-wave (bank stride 132 dwords % 64 = 4) cover all 64 banks.
#define BROW 528
#define BTILE (16 * BROW)                    // one hi or lo region (8448 B)

#if __has_builtin(__builtin_amdgcn_global_load_async_to_lds_b128) && \
    __has_builtin(__builtin_amdgcn_s_wait_asynccnt)
#define USE_ASYNC 1
#define WAIT_ASYNC(n) __builtin_amdgcn_s_wait_asynccnt(n)
typedef __attribute__((address_space(1))) v4i* gv4i_p;
typedef __attribute__((address_space(3))) v4i* lv4i_p;
#else
#define USE_ASYNC 0
#define WAIT_ASYNC(n) do { } while (0)
#endif

__device__ __forceinline__ void copy16_to_lds(const char* g, char* l) {
#if USE_ASYNC
  __builtin_amdgcn_global_load_async_to_lds_b128(
      (gv4i_p)(void*)g, (lv4i_p)(void*)l, 0, 0);
#else
  *(v4f*)l = *(const v4f*)g;   // sync fallback: global_load_b128 + ds_store_b128
#endif
}

// ---------------------------------------------------------------------------
// Prep: split emb (f32) into bf16 hi/lo pair and compute ||e_k||^2 (exact f32)
// ---------------------------------------------------------------------------
__global__ __launch_bounds__(256) void vq_prep(const float* __restrict__ emb,
                                               bf16_t* __restrict__ ehi,
                                               bf16_t* __restrict__ elo,
                                               float* __restrict__ e2) {
  __shared__ float red[256];
  const int k = blockIdx.x;
  const int d = threadIdx.x;
  float f = emb[k * DIMD + d];
  bf16_t h = (bf16_t)f;                        // RNE f32 -> bf16
  ehi[k * DIMD + d] = h;
  elo[k * DIMD + d] = (bf16_t)(f - (float)h);  // residual, next 8 mantissa bits
  red[d] = f * f;
  __syncthreads();
  for (int s = 128; s > 0; s >>= 1) {
    if (d < s) red[d] += red[d + s];
    __syncthreads();
  }
  if (d == 0) e2[k] = red[0];
}

// ---------------------------------------------------------------------------
// Main: fused (x . emb^T) GEMM via bf16x3 WMMA + per-row argmin + gather + MSE
// grid = 512 WGs x 256 threads (8 waves); each wave owns 16 rows of x.
// B k-tiles are staged into LDS once per WG (async, double-buffered).
// ---------------------------------------------------------------------------
__global__ __launch_bounds__(256) void vq_main(const float* __restrict__ x,
                                               const float* __restrict__ emb,
                                               const bf16_t* __restrict__ ehi,
                                               const bf16_t* __restrict__ elo,
                                               const float* __restrict__ e2,
                                               float* __restrict__ outq,
                                               float* __restrict__ outidx,
                                               float* __restrict__ partials) {
  __shared__ __align__(16) char Bbuf[2][2 * BTILE];   // [buf][hi | lo]

  const int lane   = threadIdx.x & 31;
  const int wave   = threadIdx.x >> 5;
  const bool hiHalf = lane >= 16;
  const int m0  = blockIdx.x * ROWS_PER_WG + wave * 16;
  const int row = m0 + (lane & 15);

  // Cooperative copy of one B k-tile (hi 8KB + lo 8KB) into an LDS buffer.
  auto issue_copy = [&](int t, int b) {
    const int k0 = t * 16;
    const char* ghi = (const char*)(ehi + (size_t)k0 * DIMD);
    const char* glo = (const char*)(elo + (size_t)k0 * DIMD);
    char* lhi = &Bbuf[b][0];
    char* llo = &Bbuf[b][BTILE];
    const int tid = threadIdx.x;
#pragma unroll
    for (int j = 0; j < 2; ++j) {
      const int c      = tid + j * 256;   // 16B chunk id, 0..511
      const int r      = c >> 5;          // 32 chunks per 512B source row
      const int within = (c & 31) * 16;
      copy16_to_lds(ghi + c * 16, lhi + r * BROW + within);
      copy16_to_lds(glo + c * 16, llo + r * BROW + within);
    }
  };

  issue_copy(0, 0);   // prefetch tile 0 while we stage A fragments

  // ---- Stage this lane's A fragments (bf16 hi/lo of its x row) in registers,
  //      laid out per the 16-bit 16x32 A-matrix VGPR layout.
  v16bf a_hi[8], a_lo[8];
  {
    const float* xr = x + (size_t)row * DIMD;
    const int dhalf = hiHalf ? 8 : 0;    // lanes>=16 hold K = +8 / +24 runs
#pragma unroll
    for (int c = 0; c < 8; ++c) {        // 8 chunks of 32 along D
      const int db = c * 32 + dhalf;
      v4f f0 = *(const v4f*)(xr + db);
      v4f f1 = *(const v4f*)(xr + db + 4);
      v4f f2 = *(const v4f*)(xr + db + 16);
      v4f f3 = *(const v4f*)(xr + db + 20);
      float fs[16];
#pragma unroll
      for (int j = 0; j < 4; ++j) {
        fs[j] = f0[j]; fs[4 + j] = f1[j]; fs[8 + j] = f2[j]; fs[12 + j] = f3[j];
      }
      v16bf ah, al;
#pragma unroll
      for (int j = 0; j < 16; ++j) {
        float f = fs[j];
        bf16_t h = (bf16_t)f;
        ah[j] = h;
        al[j] = (bf16_t)(f - (float)h);
      }
      a_hi[c] = ah;
      a_lo[c] = al;
    }
  }

  // ---- K loop: 64 tiles of 16 codebook entries, double-buffered B in LDS.
  float bestv[8];
  int   bidx[8];
#pragma unroll
  for (int v = 0; v < 8; ++v) { bestv[v] = FLT_MAX; bidx[v] = 0; }

  const int boff = hiHalf ? 8 : 0;       // B layout: lanes>=16 hold K = +8 / +24 runs
  for (int t = 0; t < DIMK / 16; ++t) {
    const int k0  = t * 16;
    const int cur = t & 1;
    if (t < DIMK / 16 - 1) {
      issue_copy(t + 1, cur ^ 1);        // prefetch next tile (other buffer)
      WAIT_ASYNC(4);                     // oldest 4 (current tile) complete
    } else {
      WAIT_ASYNC(0);
    }
    __syncthreads();                     // copies visible to all waves

    const char* bhp = &Bbuf[cur][(lane & 15) * BROW + boff * 2];
    const char* blp = bhp + BTILE;
    v8f acc0 = {0.f,0.f,0.f,0.f,0.f,0.f,0.f,0.f};  // hi*hi
    v8f acc1 = {0.f,0.f,0.f,0.f,0.f,0.f,0.f,0.f};  // lo*hi
    v8f acc2 = {0.f,0.f,0.f,0.f,0.f,0.f,0.f,0.f};  // hi*lo
#pragma unroll
    for (int c = 0; c < 8; ++c) {
      v8bf h0 = *(const v8bf*)(bhp + c * 64);
      v8bf h1 = *(const v8bf*)(bhp + c * 64 + 32);
      v8bf l0 = *(const v8bf*)(blp + c * 64);
      v8bf l1 = *(const v8bf*)(blp + c * 64 + 32);
      v16bf bhv = __builtin_shufflevector(h0, h1, 0,1,2,3,4,5,6,7,8,9,10,11,12,13,14,15);
      v16bf blv = __builtin_shufflevector(l0, l1, 0,1,2,3,4,5,6,7,8,9,10,11,12,13,14,15);
      acc0 = __builtin_amdgcn_wmma_f32_16x16x32_bf16(false, a_hi[c], false, bhv,
                                                     (short)0, acc0, false, false);
      acc1 = __builtin_amdgcn_wmma_f32_16x16x32_bf16(false, a_lo[c], false, bhv,
                                                     (short)0, acc1, false, false);
      acc2 = __builtin_amdgcn_wmma_f32_16x16x32_bf16(false, a_hi[c], false, blv,
                                                     (short)0, acc2, false, false);
    }
    const int n = k0 + (lane & 15);      // this lane's codebook column
    const float e2k = e2[n];
#pragma unroll
    for (int v = 0; v < 8; ++v) {
      float score = e2k - 2.0f * (acc0[v] + acc1[v] + acc2[v]);
      if (score < bestv[v]) { bestv[v] = score; bidx[v] = n; }
    }
    __syncthreads();                     // done reading cur before it is reused
  }

  // ---- Cross-lane argmin within each 16-lane half (xor 1,2,4,8 stays in half).
  //      C layout: VGPR v holds row M=v (lanes 0-15) / M=v+8 (lanes 16-31).
#pragma unroll
  for (int v = 0; v < 8; ++v) {
    float bv = bestv[v];
    int   bi = bidx[v];
#pragma unroll
    for (int off = 8; off >= 1; off >>= 1) {
      float ov = __shfl_xor(bv, off, 32);
      int   oi = __shfl_xor(bi, off, 32);
      if (ov < bv || (ov == bv && oi < bi)) { bv = ov; bi = oi; }  // first-occurrence
    }
    bestv[v] = bv;
    bidx[v]  = bi;
  }

  // ---- Write indices (as float, per harness output dtype).
  if ((lane & 15) == 0) {
    const int rbase = m0 + (hiHalf ? 8 : 0);
#pragma unroll
    for (int v = 0; v < 8; ++v) outidx[rbase + v] = (float)bidx[v];
  }

  // ---- Gather quantized rows (f32 emb), write output, accumulate MSE partial.
  float lsum = 0.f;
#pragma unroll
  for (int r = 0; r < 16; ++r) {
    const int idx_r = __shfl(bidx[r & 7], (r < 8) ? 0 : 16, 32);
    const int grow  = m0 + r;
    const float4* qsrc = (const float4*)(emb + (size_t)idx_r * DIMD);
    const float4* xsrc = (const float4*)(x + (size_t)grow * DIMD);
    float4* qdst = (float4*)(outq + (size_t)grow * DIMD);
#pragma unroll
    for (int tt = 0; tt < 2; ++tt) {
      const int e = lane + tt * 32;      // 64 float4 per row / 32 lanes
      float4 q = qsrc[e];
      float4 xv = xsrc[e];
      qdst[e] = q;
      float dx = q.x - xv.x, dy = q.y - xv.y, dz = q.z - xv.z, dw = q.w - xv.w;
      lsum += dx * dx + dy * dy + dz * dz + dw * dw;
    }
  }
#pragma unroll
  for (int off = 16; off >= 1; off >>= 1) lsum += __shfl_xor(lsum, off, 32);
  if (lane == 0) partials[blockIdx.x * WAVES_PER_WG + wave] = lsum;
}

// ---------------------------------------------------------------------------
// Deterministic loss reduction: 4096 per-wave partials -> scalar.
// loss = q_latent + 0.25*e_latent = 1.25 * mean((q-x)^2)
// ---------------------------------------------------------------------------
__global__ __launch_bounds__(256) void vq_loss_reduce(const float* __restrict__ partials,
                                                      float* __restrict__ loss) {
  __shared__ float red[256];
  float s = 0.f;
  for (int i = threadIdx.x; i < NUM_WAVES; i += 256) s += partials[i];
  red[threadIdx.x] = s;
  __syncthreads();
  for (int t = 128; t > 0; t >>= 1) {
    if (threadIdx.x < t) red[threadIdx.x] += red[threadIdx.x + t];
    __syncthreads();
  }
  if (threadIdx.x == 0) loss[0] = 1.25f * red[0] / 16777216.0f;
}

extern "C" void kernel_launch(void* const* d_in, const int* in_sizes, int n_in,
                              void* d_out, int out_size, void* d_ws, size_t ws_size,
                              hipStream_t stream) {
  const float* x   = (const float*)d_in[0];   // [64,1024,256] f32
  const float* emb = (const float*)d_in[1];   // [1024,256] f32

  char* ws = (char*)d_ws;
  bf16_t* ehi      = (bf16_t*)ws;                          // 512 KB
  bf16_t* elo      = (bf16_t*)(ws + 512 * 1024);           // 512 KB
  float*  e2       = (float*)(ws + 1024 * 1024);           // 4 KB
  float*  partials = (float*)(ws + 1024 * 1024 + 4096);    // 16 KB

  float* out    = (float*)d_out;
  float* outq   = out;                         // 16777216 quantized values
  float* loss   = out + 16777216;              // 1 scalar
  float* outidx = out + 16777217;              // 65536 indices (as float)

  vq_prep<<<DIMK, 256, 0, stream>>>(emb, ehi, elo, e2);
  vq_main<<<NUM_WG, 256, 0, stream>>>(x, emb, ehi, elo, e2, outq, outidx, partials);
  vq_loss_reduce<<<1, 256, 0, stream>>>(partials, loss);
}
